// dMaSIFSearchEmbed_1898375545076
// MI455X (gfx1250) — compile-verified, matrix operand order checked by reading, not run on hardware
//
#include <hip/hip_runtime.h>
#include <hip/hip_bf16.h>

#define N_PTS 1024
#define IN_CH 16
#define ORI_U 16
#define EMB 8
#define CUTS 8
#define JSPLIT 4    // j-range split factor for the O(N^2) conv kernel

typedef __attribute__((ext_vector_type(16))) _Float16 v16h;
typedef __attribute__((ext_vector_type(8)))  float    v8f;

// workspace layout per branch (in floats)
#define WS_W      0                 // N        orientation weights
#define WS_F      (N_PTS)           // N*8      features (raw then normalized in place)
#define WS_NUV    (N_PTS*9)         // N*9      local frames
#define WS_COUT   (N_PTS*18)        // N*8      conv output (atomic accumulator)
#define WS_ORAW   (N_PTS*26)        // N*8      post-oMLP (pre groupnorm)
#define WS_ST1    (N_PTS*34)        // 8        group sums/sumsq for input norm
#define WS_ST2    (N_PTS*34+8)      // 8        group sums/sumsq for output norm
#define WS_STRIDE (N_PTS*34+16)

__device__ __forceinline__ float lrelu(float x) { return x > 0.f ? x : 0.2f * x; }

// ------------------------------------- zero the atomic accumulators + stats
__global__ __launch_bounds__(256) void k_zero(float* ws) {
    int idx = blockIdx.x * blockDim.x + threadIdx.x;
    if (idx < N_PTS * EMB) {
        ws[WS_COUT + idx] = 0.f;
        ws[WS_STRIDE + WS_COUT + idx] = 0.f;
    }
    if (idx < 16) {
        ws[WS_ST1 + idx] = 0.f;
        ws[WS_STRIDE + WS_ST1 + idx] = 0.f;
    }
}

// ------------------------------------------------- per-point MLPs + stats(1)
__global__ __launch_bounds__(256) void k_prep(
    const float* __restrict__ feats,
    const float* __restrict__ osW1, const float* __restrict__ osb1,
    const float* __restrict__ osW2, const float* __restrict__ osb2,
    const float* __restrict__ inW1, const float* __restrict__ inb1,
    const float* __restrict__ inW2, const float* __restrict__ inb2,
    float* __restrict__ w_out, float* __restrict__ f_out, float* __restrict__ st1)
{
    int i = blockIdx.x * blockDim.x + threadIdx.x;
    if (i >= N_PTS) return;
    float fr[IN_CH];
#pragma unroll
    for (int c = 0; c < IN_CH; ++c) fr[c] = feats[i * IN_CH + c];

    // orientation score: 16 -> 16 (lrelu) -> 1
    float acc = osb2[0];
#pragma unroll
    for (int u = 0; u < ORI_U; ++u) {
        float h = osb1[u];
#pragma unroll
        for (int c = 0; c < IN_CH; ++c) h += fr[c] * osW1[u * IN_CH + c];
        acc += lrelu(h) * osW2[u];
    }
    w_out[i] = acc;

    // input MLP: 16 -> 8 (lrelu) -> 8 (lrelu)
    float t1[EMB], t2[EMB];
#pragma unroll
    for (int h = 0; h < EMB; ++h) {
        float s = inb1[h];
#pragma unroll
        for (int c = 0; c < IN_CH; ++c) s += fr[c] * inW1[h * IN_CH + c];
        t1[h] = lrelu(s);
    }
#pragma unroll
    for (int h = 0; h < EMB; ++h) {
        float s = inb2[h];
#pragma unroll
        for (int k = 0; k < EMB; ++k) s += t1[k] * inW2[h * EMB + k];
        t2[h] = lrelu(s);
        f_out[i * EMB + h] = t2[h];
    }
#pragma unroll
    for (int g = 0; g < 4; ++g) {
        float a = t2[2 * g], b = t2[2 * g + 1];
        atomicAdd(&st1[g], a + b);
        atomicAdd(&st1[4 + g], a * a + b * b);
    }
}

// -------------------------------------------------- load_mesh: frames (nuv)
__global__ __launch_bounds__(256) void k_mesh(
    const float* __restrict__ xyz, const float* __restrict__ nrm,
    const float* __restrict__ w, float* __restrict__ nuv)
{
    const float invR = 1.0f / 9.0f;
    int lane = threadIdx.x & 31;
    int i = blockIdx.x * 8 + (threadIdx.x >> 5);

    float px = xyz[i * 3 + 0] * invR, py = xyz[i * 3 + 1] * invR, pz = xyz[i * 3 + 2] * invR;
    float nx = nrm[i * 3 + 0], ny = nrm[i * 3 + 1], nz = nrm[i * 3 + 2];

    // tangent basis
    float s = (nz >= 0.f) ? 1.f : -1.f;
    float a = -1.f / (s + nz);
    float bb = nx * ny * a;
    float ux = 1.f + s * nx * nx * a, uy = s * bb, uz = -s * nx;
    float vx = bb, vy = s + ny * ny * a, vz = -ny;

    float ax = 0.f, ay = 0.f;
    for (int j = lane; j < N_PTS; j += 32) {
        float dx = xyz[j * 3 + 0] * invR - px;
        float dy = xyz[j * 3 + 1] * invR - py;
        float dz = xyz[j * 3 + 2] * invR - pz;
        float njx = nrm[j * 3 + 0], njy = nrm[j * 3 + 1], njz = nrm[j * 3 + 2];
        float fac = 2.f - (nx * njx + ny * njy + nz * njz);
        float rho2 = (dx * dx + dy * dy + dz * dz) * fac * fac;
        float wdw = __expf(-rho2) * w[j];
        ax += wdw * (ux * dx + uy * dy + uz * dz);
        ay += wdw * (vx * dx + vy * dy + vz * dz);
    }
#pragma unroll
    for (int off = 16; off > 0; off >>= 1) {
        ax += __shfl_down(ax, off, 32);
        ay += __shfl_down(ay, off, 32);
    }
    if (lane == 0) {
        ax += 1e-5f; ay += 1e-5f;
        float nv = sqrtf(ax * ax + ay * ay);
        nv = fmaxf(nv, 1e-12f);
        float ex = ax / nv, ey = ay / nv;
        nuv[i * 9 + 0] = nx; nuv[i * 9 + 1] = ny; nuv[i * 9 + 2] = nz;
        nuv[i * 9 + 3] = ex * ux + ey * vx;
        nuv[i * 9 + 4] = ex * uy + ey * vy;
        nuv[i * 9 + 5] = ex * uz + ey * vz;
        nuv[i * 9 + 6] = -ey * ux + ex * vx;
        nuv[i * 9 + 7] = -ey * uy + ex * vy;
        nuv[i * 9 + 8] = -ey * uz + ex * vz;
    }
}

// ---------------------------------------------------- group-norm f in place
__global__ __launch_bounds__(256) void k_normf(
    float* __restrict__ f, const float* __restrict__ st1,
    const float* __restrict__ g, const float* __restrict__ b)
{
    int i = blockIdx.x * blockDim.x + threadIdx.x;
    if (i >= N_PTS) return;
    const float invc = 1.f / (2.f * N_PTS);
#pragma unroll
    for (int h = 0; h < EMB; ++h) {
        int gg = h >> 1;
        float mean = st1[gg] * invc;
        float var = st1[4 + gg] * invc - mean * mean;
        float r = rsqrtf(var + 1e-5f);
        f[i * EMB + h] = (f[i * EMB + h] - mean) * r * g[h] + b[h];
    }
}

// ------------------------------------------- main O(N^2) conv (WMMA kernel)
// grid = (N_PTS/8) * JSPLIT blocks; each block: 8 waves, one point i per wave,
// handling (64/JSPLIT) j-tiles; results accumulate atomically into cout.
__global__ __launch_bounds__(256) void k_conv(
    const float* __restrict__ xyz, const float* __restrict__ nrm,
    const float* __restrict__ f, const float* __restrict__ nuv,
    const float* __restrict__ c1W, const float* __restrict__ c1b,
    const float* __restrict__ c2W, const float* __restrict__ c2b,
    float* __restrict__ cout)
{
    __shared__ _Float16 h1s[8][16][CUTS];   // per-wave h1 staging (A matrix rows)
    __shared__ float    wf[8][16][EMB];     // per-wave window * f[j]
    __shared__ float    accs[8][EMB];       // per-wave output accumulators

    const int lane  = threadIdx.x & 31;
    const int wv    = threadIdx.x >> 5;
    const int l16   = lane & 15;
    const int half  = lane >> 4;
    const int slice = blockIdx.x & (JSPLIT - 1);
    const int i     = (blockIdx.x / JSPLIT) * 8 + wv;
    const float INV_SR = 1.0f / (1.41421356237f * 9.0f);

    float pix = xyz[i * 3 + 0] * INV_SR;
    float piy = xyz[i * 3 + 1] * INV_SR;
    float piz = xyz[i * 3 + 2] * INV_SR;
    float nix = nrm[i * 3 + 0], niy = nrm[i * 3 + 1], niz = nrm[i * 3 + 2];
    float nv[9];
#pragma unroll
    for (int t = 0; t < 9; ++t) nv[t] = nuv[i * 9 + t];

    // B fragments: B[K=c][N=m] = c2_W[m][c], K padded 8 -> 32 with zeros.
    // B layout used here: lanes 0-15 elements 0..7 hold K=0..7 (live); all other elements are zero.
    v16h bfrag[4];
#pragma unroll
    for (int ch = 0; ch < 4; ++ch) {
#pragma unroll
        for (int e = 0; e < 16; ++e) {
            float vsrc = (half == 0 && e < CUTS) ? c2W[(ch * 16 + l16) * CUTS + e] : 0.f;
            bfrag[ch][e] = (_Float16)vsrc;
        }
    }
    if (lane < EMB) accs[wv][lane] = 0.f;
    __syncthreads();

    const int jt0 = slice * (N_PTS / 16 / JSPLIT);
    const int jt1 = jt0 + (N_PTS / 16 / JSPLIT);
    for (int jt = jt0; jt < jt1; ++jt) {
        int j = jt * 16 + l16;
        float dx = xyz[j * 3 + 0] * INV_SR - pix;
        float dy = xyz[j * 3 + 1] * INV_SR - piy;
        float dz = xyz[j * 3 + 2] * INV_SR - piz;
        if (half == 0) {
            // X = nuv_i . d ; h1 = relu(c1_W X + c1_b)  (A matrix row j)
            float X0 = nv[0] * dx + nv[1] * dy + nv[2] * dz;
            float X1 = nv[3] * dx + nv[4] * dy + nv[5] * dz;
            float X2 = nv[6] * dx + nv[7] * dy + nv[8] * dz;
#pragma unroll
            for (int c = 0; c < CUTS; ++c) {
                float h = c1W[c * 3 + 0] * X0 + c1W[c * 3 + 1] * X1 + c1W[c * 3 + 2] * X2 + c1b[c];
                h1s[wv][l16][c] = (_Float16)fmaxf(h, 0.f);
            }
        } else {
            float njx = nrm[j * 3 + 0], njy = nrm[j * 3 + 1], njz = nrm[j * 3 + 2];
            float fac = 2.f - (nix * njx + niy * njy + niz * njz);
            float d2 = (dx * dx + dy * dy + dz * dz) * fac * fac;
            float wdw = __expf(-d2);
#pragma unroll
            for (int bb = 0; bb < EMB; ++bb) wf[wv][l16][bb] = wdw * f[j * EMB + bb];
        }
        __syncthreads();

        // A fragment: 16-bit A layout; lanes 0-15 elements 0..7 = K=0..7 (live),
        // everything else (K >= 8) is zero padding.
        v16h afrag;
#pragma unroll
        for (int e = 0; e < 16; ++e) {
            _Float16 av = (_Float16)0.f;
            if (half == 0 && e < CUTS) av = h1s[wv][l16][e];
            afrag[e] = av;
        }

#pragma unroll
        for (int ch = 0; ch < 4; ++ch) {
            v8f cz = {0.f, 0.f, 0.f, 0.f, 0.f, 0.f, 0.f, 0.f};
            v8f dacc = __builtin_amdgcn_wmma_f32_16x16x32_f16(
                false, afrag, false, bfrag[ch], (short)0, cz, false, false);
            // D layout: lane -> m = ch*16 + l16 ; VGPR r -> j-row r + 8*half
            int m = ch * 16 + l16;
            float bia = c2b[m];
            int bidx = m & 7;
            float part = 0.f;
#pragma unroll
            for (int r = 0; r < 8; ++r) {
                float v = dacc[r] + bia;
                v = v > 0.f ? v : 0.f;        // relu(h2)
                part += v * wf[wv][r + 8 * half][bidx];
            }
            atomicAdd(&accs[wv][m >> 3], part);
        }
        __syncthreads();
    }

    if (lane < EMB) atomicAdd(&cout[i * EMB + lane], accs[wv][lane]);
}

// ------------------------------------------- output MLP + stats(2), per point
__global__ __launch_bounds__(256) void k_omlp(
    const float* __restrict__ cout,
    const float* __restrict__ oW1, const float* __restrict__ ob1,
    const float* __restrict__ oW2, const float* __restrict__ ob2,
    float* __restrict__ oraw, float* __restrict__ st2)
{
    int i = blockIdx.x * blockDim.x + threadIdx.x;
    if (i >= N_PTS) return;
    float ov[EMB], t1[EMB], t2[EMB];
#pragma unroll
    for (int h = 0; h < EMB; ++h) ov[h] = cout[i * EMB + h];
#pragma unroll
    for (int o = 0; o < EMB; ++o) {
        float s = ob1[o];
#pragma unroll
        for (int h = 0; h < EMB; ++h) s += ov[h] * oW1[o * EMB + h];
        t1[o] = lrelu(s);
    }
#pragma unroll
    for (int o = 0; o < EMB; ++o) {
        float s = ob2[o];
#pragma unroll
        for (int k = 0; k < EMB; ++k) s += t1[k] * oW2[o * EMB + k];
        t2[o] = lrelu(s);
        oraw[i * EMB + o] = t2[o];
    }
#pragma unroll
    for (int g = 0; g < 4; ++g) {
        float a = t2[2 * g], b = t2[2 * g + 1];
        atomicAdd(&st2[g], a + b);
        atomicAdd(&st2[4 + g], a * a + b * b);
    }
}

// ------------------------------------ output group-norm + ll/lt linears
__global__ __launch_bounds__(256) void k_final(
    const float* __restrict__ feats, const float* __restrict__ oraw,
    const float* __restrict__ st2,
    const float* __restrict__ nog, const float* __restrict__ nob,
    const float* __restrict__ llW1, const float* __restrict__ llb1,
    const float* __restrict__ llW2, const float* __restrict__ llb2,
    const float* __restrict__ ltW, const float* __restrict__ ltb,
    float* __restrict__ outp)
{
    int i = blockIdx.x * blockDim.x + threadIdx.x;
    if (i >= N_PTS) return;
    const float invc = 1.f / (2.f * N_PTS);
    float on[EMB], t1[EMB];
#pragma unroll
    for (int h = 0; h < EMB; ++h) {
        int gg = h >> 1;
        float mean = st2[gg] * invc;
        float var = st2[4 + gg] * invc - mean * mean;
        float r = rsqrtf(var + 1e-5f);
        on[h] = (oraw[i * EMB + h] - mean) * r * nog[h] + nob[h];
    }
#pragma unroll
    for (int o = 0; o < EMB; ++o) {
        float s = llb1[o];
#pragma unroll
        for (int h = 0; h < EMB; ++h) s += on[h] * llW1[o * EMB + h];
        t1[o] = fmaxf(s, 0.f);  // relu (not lrelu) in _conv_seg
    }
#pragma unroll
    for (int o = 0; o < EMB; ++o) {
        float xi = llb2[o];
#pragma unroll
        for (int k = 0; k < EMB; ++k) xi += t1[k] * llW2[o * EMB + k];
        float base = ltb[o];
#pragma unroll
        for (int c = 0; c < IN_CH; ++c) base += feats[i * IN_CH + c] * ltW[o * IN_CH + c];
        outp[i * EMB + o] = base + xi;
    }
}

extern "C" void kernel_launch(void* const* d_in, const int* in_sizes, int n_in,
                              void* d_out, int out_size, void* d_ws, size_t ws_size,
                              hipStream_t stream) {
    const float* xyz   = (const float*)d_in[0];
    const float* nrm   = (const float*)d_in[1];
    const float* feats = (const float*)d_in[2];
    float* ws  = (float*)d_ws;
    float* out = (float*)d_out;

    k_zero<<<(N_PTS * EMB + 255) / 256, 256, 0, stream>>>(ws);

    for (int b = 0; b < 2; ++b) {
        // jax tree (sorted-key) flatten order within a branch:
        const int p = 3 + b * 26;
        const float* c1W  = (const float*)d_in[p + 0];
        const float* c1b  = (const float*)d_in[p + 1];
        const float* c2W  = (const float*)d_in[p + 2];
        const float* c2b  = (const float*)d_in[p + 3];
        const float* inW1 = (const float*)d_in[p + 4];
        const float* inW2 = (const float*)d_in[p + 5];
        const float* inb1 = (const float*)d_in[p + 6];
        const float* inb2 = (const float*)d_in[p + 7];
        const float* nib  = (const float*)d_in[p + 8];
        const float* nig  = (const float*)d_in[p + 9];
        const float* nob  = (const float*)d_in[p + 10];
        const float* nog  = (const float*)d_in[p + 11];
        const float* oW1  = (const float*)d_in[p + 12];
        const float* oW2  = (const float*)d_in[p + 13];
        const float* ob1  = (const float*)d_in[p + 14];
        const float* ob2  = (const float*)d_in[p + 15];
        const float* llW1 = (const float*)d_in[p + 16];
        const float* llW2 = (const float*)d_in[p + 17];
        const float* llb1 = (const float*)d_in[p + 18];
        const float* llb2 = (const float*)d_in[p + 19];
        const float* ltW  = (const float*)d_in[p + 20];
        const float* ltb  = (const float*)d_in[p + 21];
        const float* osW1 = (const float*)d_in[p + 22];
        const float* osW2 = (const float*)d_in[p + 23];
        const float* osb1 = (const float*)d_in[p + 24];
        const float* osb2 = (const float*)d_in[p + 25];

        float* wsb  = ws + (size_t)b * WS_STRIDE;
        float* outb = out + (size_t)b * N_PTS * EMB;

        k_prep<<<N_PTS / 256, 256, 0, stream>>>(feats, osW1, osb1, osW2, osb2,
                                                inW1, inb1, inW2, inb2,
                                                wsb + WS_W, wsb + WS_F, wsb + WS_ST1);
        k_mesh<<<N_PTS / 8, 256, 0, stream>>>(xyz, nrm, wsb + WS_W, wsb + WS_NUV);
        k_normf<<<N_PTS / 256, 256, 0, stream>>>(wsb + WS_F, wsb + WS_ST1, nig, nib);
        k_conv<<<(N_PTS / 8) * JSPLIT, 256, 0, stream>>>(xyz, nrm, wsb + WS_F, wsb + WS_NUV,
                                                         c1W, c1b, c2W, c2b,
                                                         wsb + WS_COUT);
        k_omlp<<<N_PTS / 256, 256, 0, stream>>>(wsb + WS_COUT, oW1, ob1, oW2, ob2,
                                                wsb + WS_ORAW, wsb + WS_ST2);
        k_final<<<N_PTS / 256, 256, 0, stream>>>(feats, wsb + WS_ORAW, wsb + WS_ST2,
                                                 nog, nob, llW1, llb1, llW2, llb2,
                                                 ltW, ltb, outb);
    }
}
